// LSTM_38568806318131
// MI455X (gfx1250) — matrix operand order repaired and checked by reading, not compiled, not run on hardware
//
#include <hip/hip_runtime.h>

// ---------------------------------------------------------------------------
// LSTM  B=64, T=2048, I=64, H=256  (gates 4H = 1024), fp32 in/out.
//
// Strategy (MI455X / gfx1250, wave32):
//  * Recurrence is independent per batch element -> 4 persistent workgroups,
//    each owning 16 batch rows (M=16 = one WMMA tile row). Only intra-WGP
//    s_barrier per timestep; no grid/cluster sync on the serial path.
//  * Fold input GEMM into the recurrence:  gates = [x_t | h] @ [W_ih|W_hh]^T
//    with K = 64 + 256 = 320 (10 fragments of 32) -> no 512MB xg intermediate.
//  * Matmul via v_wmma_f32_16x16x32_bf16 (bf16 in, f32 accumulate).
//    Weights converted once to bf16 and pre-shuffled into per-lane B-fragment
//    order.  W (640KB) stays L2-resident; per gate one per-lane pointer is
//    kept in a stable VGPR pair (made LICM-opaque once per timestep) and all
//    10 K-fragments are addressed via the 24-bit immediate offset ->
//    global_load_b128 with zero per-load VALU and no addr-register churn.
//  * B-fragment loads use a 3-slot / distance-2 software pipeline: while the
//    WMMAs of group kf execute, group kf+1 is in flight and kf+2 is being
//    issued (~2 full WMMA groups of L2-latency cover per wave; <=24 loads
//    outstanding vs the 63-entry LOADcnt limit).  This keeps the per-wave
//    serial chain (the real critical path of the recurrence) off the L2
//    latency and forces staggered s_wait_loadcnt instead of wait-0 lockstep.
//  * Wave w owns hidden units [16w,16w+16): computes gate tiles i,f,g,o
//    (4 independent WMMA accumulation chains x 10 K-frags = 40 WMMA/step),
//    does the cell update locally (c stays in VGPRs), writes h to LDS + out.
// ---------------------------------------------------------------------------

#define B_  64
#define T_  2048
#define I_  64
#define H_  256
#define G_  1024          // 4*H
#define K_  320           // I + H
#define KF_ 10            // K / 32
#define NT_ 64            // G / 16 gate-column tiles
#define AROW_ 328         // LDS row stride in bf16 (164 DW, 164%64=36 -> 16 conflict-free rows)

typedef __attribute__((ext_vector_type(16))) __bf16 v16bf;
typedef __attribute__((ext_vector_type(8)))  __bf16 v8bf;
typedef __attribute__((ext_vector_type(8)))  float  v8f;

// global-address-space pointer -> global_load_b128 with immediate offsets
typedef const __attribute__((address_space(1))) v8bf* gv8p;

__device__ __forceinline__ v8f splat8(float v) {
    v8f r;
#pragma unroll
    for (int i = 0; i < 8; ++i) r[i] = v;
    return r;
}

__device__ __forceinline__ float fsigmoid(float x) {
    return 1.0f / (1.0f + __expf(-x));
}

#if __has_builtin(__builtin_amdgcn_tanhf)
#define TANH_(x) __builtin_amdgcn_tanhf(x)
#else
#define TANH_(x) tanhf(x)
#endif

// Load B-fragment #kf from a stable per-lane base pointer; kf*1024 bytes
// (= 64 v8bf elements) folds into the 24-bit instruction immediate.
__device__ __forceinline__ v16bf loadB(uintptr_t base, int kf) {
    gv8p p  = (gv8p)base;
    v8bf lo = p[64 * kf];
    v8bf hi = p[64 * kf + 1];
    return __builtin_shufflevector(lo, hi,
        0, 1, 2, 3, 4, 5, 6, 7, 8, 9, 10, 11, 12, 13, 14, 15);
}

// ---------------------------------------------------------------------------
// Prep: pack [W_ih | W_hh] -> bf16 B-fragments in d_ws, and bias = b_ih+b_hh.
// Block (gt,kf): 1024 bytes = 32 lanes x 16 bf16.  B-layout, contiguous K order (same as A):
//   lanes 0-15 : N = gt*16 + lane,      K = kf*32 + 0..15
//   lanes 16-31: N = gt*16 + lane - 16, K = kf*32 + 16..31
// ---------------------------------------------------------------------------
__global__ void lstm_pack(const float* __restrict__ W_ih,
                          const float* __restrict__ W_hh,
                          const float* __restrict__ b_ih,
                          const float* __restrict__ b_hh,
                          __bf16* __restrict__ wpack,
                          float* __restrict__ bias) {
    const int blk  = blockIdx.x;
    const int lane = threadIdx.x;            // 32 threads
    if (blk < NT_ * KF_) {
        const int gt = blk / KF_;
        const int kf = blk % KF_;
        const int n  = gt * 16 + (lane & 15);
        const int kb = kf * 32 + ((lane >= 16) ? 16 : 0);
        __bf16* dst = wpack + blk * 512 + lane * 16;
#pragma unroll
        for (int j = 0; j < 16; ++j) {
            const int k = kb + j;
            const float w = (k < I_) ? W_ih[n * I_ + k]
                                     : W_hh[n * H_ + (k - I_)];
            dst[j] = (__bf16)w;
        }
    } else {
        // 4 trailing blocks: bias_sum[1024]
        const int idx = (blk - NT_ * KF_) * 256 + lane * 8;
#pragma unroll
        for (int j = 0; j < 8; ++j)
            bias[idx + j] = b_ih[idx + j] + b_hh[idx + j];
    }
}

// ---------------------------------------------------------------------------
// Persistent scan kernel: grid = 4 blocks x 512 threads (16 waves).
// Block b owns batches [16b, 16b+16).  a_buf row m = [x_t(64) | h(256)] bf16.
// ---------------------------------------------------------------------------
__global__ __launch_bounds__(512, 1)
void lstm_scan(const float* __restrict__ x,
               const __bf16* __restrict__ wpack,
               const float* __restrict__ bias,
               float* __restrict__ out) {
    __shared__ __align__(16) __bf16 a_buf[16][AROW_];

    const int tid    = threadIdx.x;
    const int wave   = tid >> 5;          // 0..15
    const int lane   = tid & 31;
    const int col    = lane & 15;         // N within tile / A row (M)
    const int hiHalf = (lane >= 16) ? 1 : 0;
    const int m_base = blockIdx.x * 16;   // global batch base
    const int hu     = wave * 16;         // hidden-unit base for this wave

    // zero whole a_buf (h starts at 0; x region overwritten each step)
    for (int i = tid; i < 16 * AROW_; i += 512)
        (&a_buf[0][0])[i] = (__bf16)0.0f;

    // bias for this lane's output column in each gate (i,f,g,o)
    float bias_g[4];
#pragma unroll
    for (int g = 0; g < 4; ++g)
        bias_g[g] = bias[g * H_ + hu + col];

    // per-lane base address of this wave's B-fragments for each gate (kf=0);
    // lives in 4 stable VGPR pairs for the whole kernel
    uintptr_t wp[4];
#pragma unroll
    for (int g = 0; g < 4; ++g)
        wp[g] = (uintptr_t)(wpack
                + (size_t)((g * 16 + wave) * KF_) * 512 + lane * 16);

    v8f c_state = splat8(0.0f);
    v8f hv      = splat8(0.0f);

    __syncthreads();

    for (int t = 0; t < T_; ++t) {
        // make the 4 fragment base pointers opaque once per timestep:
        // blocks LICM of the (t-invariant) weight loads -> no scratch spills
        asm volatile("" : "+v"(wp[0]), "+v"(wp[1]), "+v"(wp[2]), "+v"(wp[3]));

        // ---- cooperative x_t load: 16 rows x 64 f32 -> bf16, 2 elems/thread
        {
            const int row = tid >> 5;                 // (tid*2)/64
            const int c0  = (tid * 2) & 63;
            const float* xs = x + ((size_t)(m_base + row) * T_ + t) * I_ + c0;
            union { __bf16 h[2]; unsigned u; } p;
            p.h[0] = (__bf16)xs[0];
            p.h[1] = (__bf16)xs[1];
            *reinterpret_cast<unsigned*>(&a_buf[row][c0]) = p.u;
            // prefetch next timestep's x row (global_prefetch_b8)
            if (t + 1 < T_) __builtin_prefetch(xs + I_, 0, 0);
        }
        __syncthreads();

        // ---- gates = bias + [x_t | h] @ Wcat^T  (4 tiles x 10 K-frags)
        v8f acc[4];
#pragma unroll
        for (int g = 0; g < 4; ++g) acc[g] = splat8(bias_g[g]);

        // 3-slot, distance-2 pipeline: compute kf | in-flight kf+1 | issue kf+2
        v16bf bbuf[3][4];
#pragma unroll
        for (int g = 0; g < 4; ++g) bbuf[0][g] = loadB(wp[g], 0);
#pragma unroll
        for (int g = 0; g < 4; ++g) bbuf[1][g] = loadB(wp[g], 1);

#pragma unroll
        for (int kf = 0; kf < KF_; ++kf) {
            // issue group kf+2 first: ~2 WMMA groups of latency cover
            if (kf + 2 < KF_) {
#pragma unroll
                for (int g = 0; g < 4; ++g)
                    bbuf[(kf + 2) % 3][g] = loadB(wp[g], kf + 2);
            }

            // A fragment (contiguous K order, matching packed B):
            //   lane<16 : halves 0-15 = K+0..15
            //   lane>=16: halves 0-15 = K+16..31
            const int koff = kf * 32 + (hiHalf ? 16 : 0);
            v8bf alo = *reinterpret_cast<const v8bf*>(&a_buf[col][koff]);
            v8bf ahi = *reinterpret_cast<const v8bf*>(&a_buf[col][koff + 8]);
            v16bf afrag = __builtin_shufflevector(alo, ahi,
                0, 1, 2, 3, 4, 5, 6, 7, 8, 9, 10, 11, 12, 13, 14, 15);

#pragma unroll
            for (int g = 0; g < 4; ++g)
                acc[g] = __builtin_amdgcn_wmma_f32_16x16x32_bf16(
                    false, afrag, false, bbuf[kf % 3][g], (short)0, acc[g],
                    false, false);
        }

        // ---- cell update (C/D layout: VGPR r, lane l -> M = r + 8*hiHalf,
        //      N = l&15).  c stays in registers; h -> LDS(bf16) + out(f32).
#pragma unroll
        for (int r = 0; r < 8; ++r) {
            const float ig = fsigmoid(acc[0][r]);
            const float fg = fsigmoid(acc[1][r]);
            const float gg = TANH_(acc[2][r]);
            const float og = fsigmoid(acc[3][r]);
            const float c  = fg * c_state[r] + ig * gg;
            c_state[r] = c;
            hv[r]      = og * TANH_(c);
        }
#pragma unroll
        for (int r = 0; r < 8; ++r) {
            const int M = r + (hiHalf ? 8 : 0);
            a_buf[M][I_ + hu + col] = (__bf16)hv[r];
            out[(((size_t)(m_base + M) * T_) + t) * H_ + hu + col] = hv[r];
        }
        __syncthreads();
    }

    // ---- final h_T, c_T (appended after lstm_out in d_out)
    const size_t fin = (size_t)B_ * T_ * H_;
#pragma unroll
    for (int r = 0; r < 8; ++r) {
        const int M = r + (hiHalf ? 8 : 0);
        out[fin + (size_t)(m_base + M) * H_ + hu + col]                   = hv[r];
        out[fin + (size_t)B_ * H_ + (size_t)(m_base + M) * H_ + hu + col] = c_state[r];
    }
}

// ---------------------------------------------------------------------------
extern "C" void kernel_launch(void* const* d_in, const int* in_sizes, int n_in,
                              void* d_out, int out_size, void* d_ws, size_t ws_size,
                              hipStream_t stream) {
    (void)in_sizes; (void)n_in; (void)out_size; (void)ws_size;
    const float* x    = (const float*)d_in[0];
    const float* W_ih = (const float*)d_in[1];
    const float* W_hh = (const float*)d_in[2];
    const float* b_ih = (const float*)d_in[3];
    const float* b_hh = (const float*)d_in[4];
    float* out = (float*)d_out;

    __bf16* wpack = (__bf16*)d_ws;                                    // 640 KB
    float*  bias  = (float*)((char*)d_ws + (size_t)NT_ * KF_ * 1024); // +4 KB

    lstm_pack<<<NT_ * KF_ + 4, 32, 0, stream>>>(W_ih, W_hh, b_ih, b_hh,
                                                wpack, bias);
    lstm_scan<<<4, 512, 0, stream>>>(x, wpack, bias, out);
}
